// QuaternionLinear_24223615550298
// MI455X (gfx1250) — matrix operand order repaired and checked by reading, not compiled, not run on hardware
//
#include <hip/hip_runtime.h>

typedef __attribute__((ext_vector_type(16))) __bf16 v16bf;
typedef __attribute__((ext_vector_type(8)))  float  v8f;

typedef int v4i __attribute__((vector_size(16)));
typedef __attribute__((address_space(1))) v4i gas_v4i;   // global int4
typedef __attribute__((address_space(3))) v4i las_v4i;   // LDS int4

#define ASYNC_B128(gp, lp) \
    __builtin_amdgcn_global_load_async_to_lds_b128( \
        (gas_v4i*)(void*)(gp), (las_v4i*)(void*)(lp), 0, 0)

#define IN_F   1024
#define OUT_F  1024
#define QDIM   256
#define M_TOT  (8 * 4096)

#define BM 256
#define BN 128
#define BK 32
#define A_STRIDEF 36   // fp32 elems per LDS A row (144B: 16B aligned, 36-dword bank stride)
#define B_STRIDE  136  // bf16 elems per LDS B row (272B: 16B aligned, 68-dword bank stride)

#if __has_builtin(__builtin_amdgcn_global_load_async_to_lds_b128)
#define HAVE_ASYNC_LDS 1
#else
#define HAVE_ASYNC_LDS 0
#endif

static __device__ __forceinline__ void wait_asynccnt0() {
#if __has_builtin(__builtin_amdgcn_s_wait_asynccnt)
    __builtin_amdgcn_s_wait_asynccnt(0);
#else
    asm volatile("s_wait_asynccnt 0x0" ::: "memory");
#endif
}

// ---------------------------------------------------------------------------
// Kernel 1: Wt[k][n] = bf16( W[n][k] ), W assembled from quaternion blocks.
// ---------------------------------------------------------------------------
__global__ __launch_bounds__(256)
void qlin_build_wt(const float* __restrict__ R, const float* __restrict__ I,
                   const float* __restrict__ J, const float* __restrict__ K,
                   __bf16* __restrict__ Wt) {
    int idx = blockIdx.x * 256 + threadIdx.x;      // 0 .. 1M-1
    int k = idx >> 10;
    int n = idx & (OUT_F - 1);
    int a = n >> 8, ro = n & 255;
    int b = k >> 8, ci = k & 255;
    int e = a * 4 + b;

    const int   sel[16] = {0,1,2,3, 1,0,3,2, 2,3,0,1, 3,2,1,0};
    const float sgn[16] = { 1.f,-1.f,-1.f,-1.f,
                            1.f, 1.f,-1.f, 1.f,
                            1.f, 1.f, 1.f,-1.f,
                            1.f,-1.f, 1.f, 1.f };

    int s = sel[e];
    const float* src = (s == 0) ? R : (s == 1) ? I : (s == 2) ? J : K;
    float v = src[ro * QDIM + ci] * sgn[e];
    Wt[(size_t)k * OUT_F + n] = (__bf16)v;
}

// ---------------------------------------------------------------------------
// Kernel 2: out[m,n] = sum_k x[m,k] * Wt[k,n] + bias[n]
//   256 threads (8 wave32), 256x128 block tile, BK=32.
//   A staged fp32, B staged bf16 via GLOBAL_LOAD_ASYNC_TO_LDS_B128 (ASYNCcnt),
//   LDS double-buffered, 1 barrier per K-step.
//   Each wave: 32 rows x 128 cols = 16 wmma per K-step, B frag reused x2,
//   B fragment registers double-buffered to hide LDS latency.
// ---------------------------------------------------------------------------
__global__ __launch_bounds__(256)
void qlin_gemm(const float* __restrict__ X, const __bf16* __restrict__ Wt,
               const float* __restrict__ bias, float* __restrict__ Out) {
    __shared__ float  AldsF[2][BM * A_STRIDEF];  // 2 * 36864 B
    __shared__ __bf16 Blds [2][BK * B_STRIDE];   // 2 *  8704 B

    const int t    = threadIdx.x;
    const int wave = t >> 5;
    const int lane = t & 31;
    const int half = lane >> 4;
    const int l16  = lane & 15;

    const int m0 = blockIdx.y * BM;
    const int n0 = blockIdx.x * BN;

    // cooperative staging indexing (branch-free)
    const int a_c = (t & 7) * 4;     // fp32 col within BK
    const int a_r = t >> 3;          // 0..31, +i*32 covers BM=256 (8 steps)
    const int b_c = (t & 7) * 16;    // bf16 col within BN
    const int b_r = t >> 3;          // 0..31 = full BK

    const float*  aG = X  + (size_t)(m0 + a_r) * IN_F + a_c;   // + kpos + i*32*IN_F
    const __bf16* bG = Wt + (size_t)b_r * OUT_F + n0 + b_c;    // + kpos*OUT_F

    // WMMA fragment indexing: wave owns rows [wave*32, wave*32+32)
    const int a_row = wave * 32 + l16;   // A frag 0 row; frag 1 at +16
    const int a_kb  = half * 8;          // A: K segment base (fp32 elems)
    const int b_row = l16 + half * 16;   // B: lane k

    v8f zero = {};
    v8f acc[2][8];
#pragma unroll
    for (int r = 0; r < 2; ++r)
#pragma unroll
        for (int j = 0; j < 8; ++j) acc[r][j] = zero;

#if HAVE_ASYNC_LDS
    auto issue = [&](int kpos, int buf) {
#pragma unroll
        for (int i = 0; i < 8; ++i) {
            const float* g = aG + kpos + (size_t)i * 32 * IN_F;
            float*       l = &AldsF[buf][(a_r + i * 32) * A_STRIDEF + a_c];
            ASYNC_B128(g, l);
        }
        const __bf16* gb = bG + (size_t)kpos * OUT_F;
        __bf16*       lb = &Blds[buf][b_r * B_STRIDE + b_c];
        ASYNC_B128(gb, lb);
        ASYNC_B128(gb + 8, lb + 8);
    };
#else
    // fallback: synchronous register staging
    auto issue = [&](int kpos, int buf) {
#pragma unroll
        for (int i = 0; i < 8; ++i) {
            float4 f = *(const float4*)(aG + kpos + (size_t)i * 32 * IN_F);
            float* dst = &AldsF[buf][(a_r + i * 32) * A_STRIDEF + a_c];
            *(float4*)dst = f;
        }
        const uint4* bsrc = (const uint4*)(bG + (size_t)kpos * OUT_F);
        uint4* dstB = (uint4*)&Blds[buf][b_r * B_STRIDE + b_c];
        dstB[0] = bsrc[0];
        dstB[1] = bsrc[1];
    };
#endif

    auto compute = [&](int buf) {
        // ---- issue ALL raw A segment loads first (clustered, one wait) ----
        const float* ap  = &AldsF[buf][a_row * A_STRIDEF + a_kb];
        const float* ap1 = ap + 16 * A_STRIDEF;
        float4 fa0 = *(const float4*)(ap);
        float4 fa1 = *(const float4*)(ap + 4);
        float4 fa2 = *(const float4*)(ap + 16);
        float4 fa3 = *(const float4*)(ap + 20);
        float4 fa4 = *(const float4*)(ap1);
        float4 fa5 = *(const float4*)(ap1 + 4);
        float4 fa6 = *(const float4*)(ap1 + 16);
        float4 fa7 = *(const float4*)(ap1 + 20);

        // ---- issue B fragment j=0 before converting A ----
        const __bf16* bp = &Blds[buf][half * 8 * B_STRIDE + l16];
        union Bf { v16bf v; uint4 q[2]; __bf16 e[16]; } bf[2];
#pragma unroll
        for (int s = 0; s < 16; ++s) bf[0].e[s] = bp[((s < 8) ? s : s + 8) * B_STRIDE];

        // ---- convert A (fa* -> v_cvt_pk_bf16_f32) ----
        union { v16bf v; __bf16 e[16]; } af0, af1;
        af0.e[0]  = (__bf16)fa0.x; af0.e[1]  = (__bf16)fa0.y;
        af0.e[2]  = (__bf16)fa0.z; af0.e[3]  = (__bf16)fa0.w;
        af0.e[4]  = (__bf16)fa1.x; af0.e[5]  = (__bf16)fa1.y;
        af0.e[6]  = (__bf16)fa1.z; af0.e[7]  = (__bf16)fa1.w;
        af0.e[8]  = (__bf16)fa2.x; af0.e[9]  = (__bf16)fa2.y;
        af0.e[10] = (__bf16)fa2.z; af0.e[11] = (__bf16)fa2.w;
        af0.e[12] = (__bf16)fa3.x; af0.e[13] = (__bf16)fa3.y;
        af0.e[14] = (__bf16)fa3.z; af0.e[15] = (__bf16)fa3.w;
        af1.e[0]  = (__bf16)fa4.x; af1.e[1]  = (__bf16)fa4.y;
        af1.e[2]  = (__bf16)fa4.z; af1.e[3]  = (__bf16)fa4.w;
        af1.e[4]  = (__bf16)fa5.x; af1.e[5]  = (__bf16)fa5.y;
        af1.e[6]  = (__bf16)fa5.z; af1.e[7]  = (__bf16)fa5.w;
        af1.e[8]  = (__bf16)fa6.x; af1.e[9]  = (__bf16)fa6.y;
        af1.e[10] = (__bf16)fa6.z; af1.e[11] = (__bf16)fa6.w;
        af1.e[12] = (__bf16)fa7.x; af1.e[13] = (__bf16)fa7.y;
        af1.e[14] = (__bf16)fa7.z; af1.e[15] = (__bf16)fa7.w;

        // ---- j loop: double-buffered B fragment registers ----
#pragma unroll
        for (int j = 0; j < 8; ++j) {
            if (j < 7) {   // prefetch j+1 before consuming j
#pragma unroll
                for (int s = 0; s < 16; ++s)
                    bf[(j + 1) & 1].e[s] = bp[((s < 8) ? s : s + 8) * B_STRIDE + (j + 1) * 16];
            }
            acc[0][j] = __builtin_amdgcn_wmma_f32_16x16x32_bf16(
                false, af0.v, false, bf[j & 1].v, (short)0, acc[0][j], false, false);
            acc[1][j] = __builtin_amdgcn_wmma_f32_16x16x32_bf16(
                false, af1.v, false, bf[j & 1].v, (short)0, acc[1][j], false, false);
        }
    };

    // ---- pipeline prologue: tile 0 -> LDS buf 0 ----
    issue(0, 0);
    wait_asynccnt0();
    __syncthreads();

    // ---- main loop: unrolled x2 so buffer index is compile-time ----
    for (int kk = 0; kk < IN_F; kk += 2 * BK) {
        const int k1 = kk + BK;                                        // always valid
        const int k2 = (kk + 2 * BK < IN_F) ? kk + 2 * BK : IN_F - BK; // clamped tail

        issue(k1, 1);       // async copies in flight into buf1...
        compute(0);         // ...hidden under ds_load_b128 + 16 wmma on buf0
        wait_asynccnt0();
        __syncthreads();

        issue(k2, 0);
        compute(1);
        wait_asynccnt0();
        __syncthreads();
    }

    // ---- epilogue: n = n0 + 16j + l16, m = m0 + 32w + 16r + 8*half + p ----
#pragma unroll
    for (int r = 0; r < 2; ++r) {
        const int mbase = m0 + wave * 32 + r * 16 + half * 8;
#pragma unroll
        for (int j = 0; j < 8; ++j) {
            const int n = n0 + j * 16 + l16;
            const float bv = bias[n];
#pragma unroll
            for (int p = 0; p < 8; ++p) {
                Out[(size_t)(mbase + p) * OUT_F + n] = acc[r][j][p] + bv;
            }
        }
    }
}

// ---------------------------------------------------------------------------
extern "C" void kernel_launch(void* const* d_in, const int* in_sizes, int n_in,
                              void* d_out, int out_size, void* d_ws, size_t ws_size,
                              hipStream_t stream) {
    const float* X    = (const float*)d_in[0];
    const float* R    = (const float*)d_in[1];
    const float* I    = (const float*)d_in[2];
    const float* J    = (const float*)d_in[3];
    const float* K    = (const float*)d_in[4];
    const float* bias = (const float*)d_in[5];
    float* Out = (float*)d_out;

    __bf16* Wt = (__bf16*)d_ws;   // 2 MB scratch

    qlin_build_wt<<<(IN_F * OUT_F) / 256, 256, 0, stream>>>(R, I, J, K, Wt);

    dim3 grid(OUT_F / BN, M_TOT / BM);   // (8, 128)
    qlin_gemm<<<grid, 256, 0, stream>>>(X, Wt, bias, Out);
}
